// SparseBottleneck_14053132992599
// MI455X (gfx1250) — matrix operand order repaired and checked by reading, compiled, not run on hardware
//
#include <hip/hip_runtime.h>
#include <hip/hip_bf16.h>

// ---------------------------------------------------------------------------
// Sparse bottleneck:  ReLU(BN(x@W1)) -> 3x3 gather-GEMM -> BN/ReLU -> @W3 ->
// BN -> +x -> ReLU.   bf16 WMMA (f32 accum), stats-pass + recompute-pass BN.
// ---------------------------------------------------------------------------

typedef __attribute__((ext_vector_type(16))) __bf16 v16bf;
typedef __attribute__((ext_vector_type(8)))  float  v8f;

union BF16Frag { v16bf v; uint4 u[2]; };

static_assert(sizeof(v16bf) == 32, "v16bf size");

__device__ __forceinline__ v8f wmma_bf16(v16bf a, v16bf b, v8f c) {
  return __builtin_amdgcn_wmma_f32_16x16x32_bf16(false, a, false, b, (short)0, c,
                                                 false, false);
}

// B fragment from LDS: packed layout [tile][lane][16 elems] (32 B per lane).
__device__ __forceinline__ v16bf load_b_lds(const __bf16* s, int tile, int lane) {
  BF16Frag f;
  const uint4* p = (const uint4*)(s + (((tile << 5) + lane) << 4));
  f.u[0] = p[0];
  f.u[1] = p[1];
  return f.v;
}

// A fragment (16x32 bf16, M x K) from an f32 row-major row pointer.
// lane<16 (hi8=0): elems 0..7 = K kb+0..7, elems 8..15 = K kb+16..23
// lane>=16(hi8=8): elems 0..7 = K kb+8..15, elems 8..15 = K kb+24..31
__device__ __forceinline__ v16bf load_a_f32(const float* rowp, int kb, int hi8) {
  const float* p0 = rowp + kb + hi8;
  const float* p1 = p0 + 16;
  float4 f0 = *(const float4*)p0;
  float4 f1 = *(const float4*)(p0 + 4);
  float4 f2 = *(const float4*)p1;
  float4 f3 = *(const float4*)(p1 + 4);
  v16bf a;
  a[0]  = (__bf16)f0.x; a[1]  = (__bf16)f0.y; a[2]  = (__bf16)f0.z; a[3]  = (__bf16)f0.w;
  a[4]  = (__bf16)f1.x; a[5]  = (__bf16)f1.y; a[6]  = (__bf16)f1.z; a[7]  = (__bf16)f1.w;
  a[8]  = (__bf16)f2.x; a[9]  = (__bf16)f2.y; a[10] = (__bf16)f2.z; a[11] = (__bf16)f2.w;
  a[12] = (__bf16)f3.x; a[13] = (__bf16)f3.y; a[14] = (__bf16)f3.z; a[15] = (__bf16)f3.w;
  return a;
}

// Same A fragment but from a bf16 row (two 16-byte loads).
__device__ __forceinline__ v16bf load_a_bf16(const __bf16* rowp, int kb, int hi8) {
  BF16Frag f;
  f.u[0] = *(const uint4*)(rowp + kb + hi8);
  f.u[1] = *(const uint4*)(rowp + kb + 16 + hi8);
  return f.v;
}

// ---------------------------------------------------------------------------
// Weight repack into per-lane-contiguous WMMA B fragments (bf16).
// Wp[((tk*NT+tn)*32 + lane)*16 + e] = W[k, n],
//   n = tn*16 + (lane&15),  k = tk*32 + ((lane>>4)<<3) + (e&7) + ((e>>3)<<4)
// ---------------------------------------------------------------------------
__global__ void pack_b_k(const float* __restrict__ W, __bf16* __restrict__ Wp,
                         int Ncols, int total) {
  int t = blockIdx.x * blockDim.x + threadIdx.x;
  if (t >= total) return;
  int e    = t & 15;
  int l    = (t >> 4) & 31;
  int tile = t >> 9;
  int NT = Ncols >> 4;
  int tk = tile / NT, tn = tile - tk * NT;
  int n = (l & 15) + (tn << 4);
  int k = (tk << 5) + ((l >> 4) << 3) + (e & 7) + ((e >> 3) << 4);
  Wp[t] = (__bf16)W[(size_t)k * Ncols + n];
}

__global__ void zero_k(float* p, int n) {
  int i = blockIdx.x * blockDim.x + threadIdx.x;
  if (i < n) p[i] = 0.f;
}

__global__ void bnfin_k(const float* __restrict__ sum, const float* __restrict__ ss,
                        const float* __restrict__ gamma, const float* __restrict__ beta,
                        float* __restrict__ scale, float* __restrict__ shift,
                        int C, float invN) {
  int c = blockIdx.x * blockDim.x + threadIdx.x;
  if (c >= C) return;
  float m = sum[c] * invN;
  float v = fmaxf(ss[c] * invN - m * m, 0.f);
  float s = gamma[c] * rsqrtf(v + 1e-5f);
  scale[c] = s;
  shift[c] = beta[c] - m * s;
}

// ---------------------------------------------------------------------------
// GEMM1: [N,256] f32 x W1p -> 64 ch.  MODE 0: stats only. MODE 1: BN+ReLU ->
// bf16 h1.  Block = 8 waves, each wave = 16 rows x 64 cols (4 accumulators).
// ---------------------------------------------------------------------------
template <int MODE>
__global__ __launch_bounds__(256) void gemm1_k(
    const float* __restrict__ feat, const __bf16* __restrict__ Wp,
    float* __restrict__ gsum, float* __restrict__ gss,
    const float* __restrict__ scale, const float* __restrict__ shift,
    __bf16* __restrict__ hout, int N) {
  __shared__ __align__(16) __bf16 sW[16384];  // 8 ktiles * 4 ntiles * 512 = 32KB
  __shared__ float sSum[64], sSS[64];
  int tid = threadIdx.x;
  {
    const uint4* src = (const uint4*)Wp;
    uint4* dst = (uint4*)sW;
#pragma unroll
    for (int i = 0; i < 8; ++i) dst[tid + 256 * i] = src[tid + 256 * i];
  }
  if (MODE == 0 && tid < 64) { sSum[tid] = 0.f; sSS[tid] = 0.f; }
  __syncthreads();

  int lane = tid & 31, wave = tid >> 5;
  int hi = lane >> 4, hi8 = hi << 3, ncol = lane & 15;
  int tile_m = blockIdx.x * 128 + wave * 16;
  int mrow = tile_m + ncol;
  int mld = mrow < N ? mrow : N - 1;
  const float* rowp = feat + (size_t)mld * 256;

  v8f acc[4];
#pragma unroll
  for (int t = 0; t < 4; ++t)
#pragma unroll
    for (int e = 0; e < 8; ++e) acc[t][e] = 0.f;

#pragma unroll
  for (int tk = 0; tk < 8; ++tk) {
    v16bf a = load_a_f32(rowp, tk * 32, hi8);
#pragma unroll
    for (int tn = 0; tn < 4; ++tn) {
      v16bf b = load_b_lds(sW, tk * 4 + tn, lane);
      acc[tn] = wmma_bf16(a, b, acc[tn]);
    }
  }

  if (MODE == 0) {
#pragma unroll
    for (int tn = 0; tn < 4; ++tn) {
      float s = 0.f, s2 = 0.f;
#pragma unroll
      for (int r = 0; r < 8; ++r) {
        int m = tile_m + r + hi * 8;
        float v = (m < N) ? acc[tn][r] : 0.f;
        s += v; s2 += v * v;
      }
      atomicAdd(&sSum[tn * 16 + ncol], s);
      atomicAdd(&sSS[tn * 16 + ncol], s2);
    }
    __syncthreads();
    if (tid < 64) { atomicAdd(&gsum[tid], sSum[tid]); atomicAdd(&gss[tid], sSS[tid]); }
  } else {
#pragma unroll
    for (int tn = 0; tn < 4; ++tn) {
      int ch = tn * 16 + ncol;
      float sc = scale[ch], sh = shift[ch];
#pragma unroll
      for (int r = 0; r < 8; ++r) {
        int m = tile_m + r + hi * 8;
        if (m < N) {
          float v = fmaxf(acc[tn][r] * sc + sh, 0.f);
          hout[(size_t)m * 64 + ch] = (__bf16)v;
        }
      }
    }
  }
}

// ---------------------------------------------------------------------------
// GEMM2: gather 9 neighbors of h1 (bf16, L2-resident) -> K=576 GEMM with W2p.
// W2p staged through LDS in two 36KB phases.
// ---------------------------------------------------------------------------
template <int MODE>
__global__ __launch_bounds__(256) void gemm2_k(
    const __bf16* __restrict__ h1, const int* __restrict__ nbr,
    const __bf16* __restrict__ Wp,
    float* __restrict__ gsum, float* __restrict__ gss,
    const float* __restrict__ scale, const float* __restrict__ shift,
    __bf16* __restrict__ hout, int N) {
  __shared__ __align__(16) __bf16 sW[18432];  // 9 ktiles * 4 ntiles * 512 = 36KB
  __shared__ float sSum[64], sSS[64];
  int tid = threadIdx.x;
  if (MODE == 0 && tid < 64) { sSum[tid] = 0.f; sSS[tid] = 0.f; }

  int lane = tid & 31, wave = tid >> 5;
  int hi = lane >> 4, hi8 = hi << 3, ncol = lane & 15;
  int tile_m = blockIdx.x * 128 + wave * 16;
  int mrow = tile_m + ncol;
  int mld = mrow < N ? mrow : N - 1;
  const int* ip = nbr + (size_t)mld * 9;
  int id[9];
#pragma unroll
  for (int j = 0; j < 9; ++j) id[j] = ip[j];

  v8f acc[4];
#pragma unroll
  for (int t = 0; t < 4; ++t)
#pragma unroll
    for (int e = 0; e < 8; ++e) acc[t][e] = 0.f;

  for (int p = 0; p < 2; ++p) {
    __syncthreads();
    {
      const uint4* src = (const uint4*)(Wp + p * 18432);
      uint4* dst = (uint4*)sW;
#pragma unroll
      for (int i = 0; i < 9; ++i) dst[tid + 256 * i] = src[tid + 256 * i];
    }
    __syncthreads();
#pragma unroll
    for (int tkl = 0; tkl < 9; ++tkl) {
      int tk = p * 9 + tkl;
      const __bf16* rowp = h1 + (size_t)id[tk >> 1] * 64;
      v16bf a = load_a_bf16(rowp, (tk & 1) * 32, hi8);
#pragma unroll
      for (int tn = 0; tn < 4; ++tn) {
        v16bf b = load_b_lds(sW, tkl * 4 + tn, lane);
        acc[tn] = wmma_bf16(a, b, acc[tn]);
      }
    }
  }

  if (MODE == 0) {
#pragma unroll
    for (int tn = 0; tn < 4; ++tn) {
      float s = 0.f, s2 = 0.f;
#pragma unroll
      for (int r = 0; r < 8; ++r) {
        int m = tile_m + r + hi * 8;
        float v = (m < N) ? acc[tn][r] : 0.f;
        s += v; s2 += v * v;
      }
      atomicAdd(&sSum[tn * 16 + ncol], s);
      atomicAdd(&sSS[tn * 16 + ncol], s2);
    }
    __syncthreads();
    if (tid < 64) { atomicAdd(&gsum[tid], sSum[tid]); atomicAdd(&gss[tid], sSS[tid]); }
  } else {
#pragma unroll
    for (int tn = 0; tn < 4; ++tn) {
      int ch = tn * 16 + ncol;
      float sc = scale[ch], sh = shift[ch];
#pragma unroll
      for (int r = 0; r < 8; ++r) {
        int m = tile_m + r + hi * 8;
        if (m < N) {
          float v = fmaxf(acc[tn][r] * sc + sh, 0.f);
          hout[(size_t)m * 64 + ch] = (__bf16)v;
        }
      }
    }
  }
}

// ---------------------------------------------------------------------------
// GEMM3: h2 [N,64] bf16 x W3p -> 256 ch.  blockIdx.y selects 64-ch column
// group.  MODE 1 fuses BN + residual(features) + ReLU -> d_out (f32).
// ---------------------------------------------------------------------------
template <int MODE>
__global__ __launch_bounds__(256) void gemm3_k(
    const __bf16* __restrict__ h2, const float* __restrict__ feat,
    const __bf16* __restrict__ Wp,
    float* __restrict__ gsum, float* __restrict__ gss,
    const float* __restrict__ scale, const float* __restrict__ shift,
    float* __restrict__ out, int N) {
  __shared__ __align__(16) __bf16 sW[4096];  // 2 ktiles * 4 ntiles * 512 = 8KB
  __shared__ float sSum[64], sSS[64];
  int tid = threadIdx.x;
  int by = blockIdx.y;  // 0..3 column group (64 channels each)
  {
#pragma unroll
    for (int c = 0; c < 2; ++c) {
      const uint4* src = (const uint4*)(Wp + (c * 16 + by * 4) * 512);
      uint4* dst = (uint4*)(sW + c * 2048);
      dst[tid] = src[tid];  // 2048 elems = 256 uint4 per chunk; tid<256
    }
  }
  if (MODE == 0 && tid < 64) { sSum[tid] = 0.f; sSS[tid] = 0.f; }
  __syncthreads();

  int lane = tid & 31, wave = tid >> 5;
  int hi = lane >> 4, hi8 = hi << 3, ncol = lane & 15;
  int tile_m = blockIdx.x * 128 + wave * 16;
  int mrow = tile_m + ncol;
  int mld = mrow < N ? mrow : N - 1;
  const __bf16* rowp = h2 + (size_t)mld * 64;

  v8f acc[4];
#pragma unroll
  for (int t = 0; t < 4; ++t)
#pragma unroll
    for (int e = 0; e < 8; ++e) acc[t][e] = 0.f;

#pragma unroll
  for (int tk = 0; tk < 2; ++tk) {
    v16bf a = load_a_bf16(rowp, tk * 32, hi8);
#pragma unroll
    for (int tn = 0; tn < 4; ++tn) {
      v16bf b = load_b_lds(sW, tk * 4 + tn, lane);
      acc[tn] = wmma_bf16(a, b, acc[tn]);
    }
  }

  if (MODE == 0) {
#pragma unroll
    for (int tn = 0; tn < 4; ++tn) {
      float s = 0.f, s2 = 0.f;
#pragma unroll
      for (int r = 0; r < 8; ++r) {
        int m = tile_m + r + hi * 8;
        float v = (m < N) ? acc[tn][r] : 0.f;
        s += v; s2 += v * v;
      }
      atomicAdd(&sSum[tn * 16 + ncol], s);
      atomicAdd(&sSS[tn * 16 + ncol], s2);
    }
    __syncthreads();
    if (tid < 64) {
      atomicAdd(&gsum[by * 64 + tid], sSum[tid]);
      atomicAdd(&gss[by * 64 + tid], sSS[tid]);
    }
  } else {
#pragma unroll
    for (int tn = 0; tn < 4; ++tn) {
      int ch = by * 64 + tn * 16 + ncol;
      float sc = scale[ch], sh = shift[ch];
#pragma unroll
      for (int r = 0; r < 8; ++r) {
        int m = tile_m + r + hi * 8;
        if (m < N) {
          size_t o = (size_t)m * 256 + ch;
          out[o] = fmaxf(acc[tn][r] * sc + sh + feat[o], 0.f);
        }
      }
    }
  }
}

// ---------------------------------------------------------------------------
extern "C" void kernel_launch(void* const* d_in, const int* in_sizes, int n_in,
                              void* d_out, int out_size, void* d_ws, size_t ws_size,
                              hipStream_t stream) {
  const float* feat = (const float*)d_in[0];
  const int*   nbr  = (const int*)d_in[1];
  const float* W1   = (const float*)d_in[2];
  const float* g1   = (const float*)d_in[3];
  const float* b1   = (const float*)d_in[4];
  const float* W2   = (const float*)d_in[5];
  const float* g2   = (const float*)d_in[6];
  const float* b2   = (const float*)d_in[7];
  const float* W3   = (const float*)d_in[8];
  const float* g3v  = (const float*)d_in[9];
  const float* b3v  = (const float*)d_in[10];

  int N = in_sizes[0] / 256;

  char* ws = (char*)d_ws;
  __bf16* W1p = (__bf16*)(ws);                       // 32768 B
  __bf16* W2p = (__bf16*)(ws + 32768);               // 73728 B
  __bf16* W3p = (__bf16*)(ws + 32768 + 73728);       // 32768 B
  float*  st  = (float*)(ws + 139264);               // stats: 1536 floats
  float* sum1 = st +    0; float* ss1 = st +   64;
  float* sum2 = st +  128; float* ss2 = st +  192;
  float* sum3 = st +  256; float* ss3 = st +  512;
  float* sc1  = st +  768; float* sh1 = st +  832;
  float* sc2  = st +  896; float* sh2 = st +  960;
  float* sc3  = st + 1024; float* sh3 = st + 1280;
  __bf16* h1 = (__bf16*)(ws + 147456);
  __bf16* h2 = (__bf16*)(ws + 147456 + (size_t)N * 128);

  float invN = 1.0f / (float)N;
  int nb = (N + 127) >> 7;
  dim3 grid3(nb, 4);

  zero_k<<<3, 256, 0, stream>>>(st, 768);
  pack_b_k<<<64, 256, 0, stream>>>(W1, W1p, 64, 256 * 64);
  pack_b_k<<<144, 256, 0, stream>>>(W2, W2p, 64, 576 * 64);
  pack_b_k<<<64, 256, 0, stream>>>(W3, W3p, 256, 64 * 256);

  gemm1_k<0><<<nb, 256, 0, stream>>>(feat, W1p, sum1, ss1, nullptr, nullptr, nullptr, N);
  bnfin_k<<<1, 64, 0, stream>>>(sum1, ss1, g1, b1, sc1, sh1, 64, invN);
  gemm1_k<1><<<nb, 256, 0, stream>>>(feat, W1p, nullptr, nullptr, sc1, sh1, h1, N);

  gemm2_k<0><<<nb, 256, 0, stream>>>(h1, nbr, W2p, sum2, ss2, nullptr, nullptr, nullptr, N);
  bnfin_k<<<1, 64, 0, stream>>>(sum2, ss2, g2, b2, sc2, sh2, 64, invN);
  gemm2_k<1><<<nb, 256, 0, stream>>>(h1, nbr, W2p, nullptr, nullptr, sc2, sh2, h2, N);

  gemm3_k<0><<<grid3, 256, 0, stream>>>(h2, feat, W3p, sum3, ss3, nullptr, nullptr, nullptr, N);
  bnfin_k<<<1, 256, 0, stream>>>(sum3, ss3, g3v, b3v, sc3, sh3, 256, invN);
  gemm3_k<1><<<grid3, 256, 0, stream>>>(h2, feat, W3p, nullptr, nullptr, sc3, sh3,
                                        (float*)d_out, N);
}
